// NodeAdaptiveExit_77129022702250
// MI455X (gfx1250) — matrix operand order repaired and checked by reading, not compiled, not run on hardware
//
#include <hip/hip_runtime.h>
#include <hip/hip_bf16.h>
#include <math.h>

typedef _Float16 f16;
typedef __attribute__((ext_vector_type(16))) _Float16 v16h;
typedef __attribute__((ext_vector_type(8)))  _Float16 v8h;
typedef __attribute__((ext_vector_type(8)))  float    v8f;

#define H_DIM   128
#define CH_DIM  64
#define NLAYERS 4
#define TAU0_C  1.0f
#define LN_EPS_C 1e-5f

#if defined(__has_builtin)
#if __has_builtin(__builtin_amdgcn_tensor_load_to_lds) && __has_builtin(__builtin_amdgcn_s_wait_tensorcnt)
#define USE_TDM 1
#endif
#endif
#ifndef USE_TDM
#define USE_TDM 0
#endif

// ---------------------------------------------------------------- helpers

__device__ __forceinline__ float gelu_exact(float x) {
    return 0.5f * x * (1.0f + erff(x * 0.7071067811865476f));
}

__device__ __forceinline__ unsigned hash3(unsigned a, unsigned b, unsigned c) {
    unsigned h = a * 0x9E3779B1u + b * 0x85EBCA77u + c * 0xC2B2AE3Du + 0x27D4EB2Fu;
    h ^= h >> 16; h *= 0x7FEB352Du;
    h ^= h >> 15; h *= 0x846CA68Bu;
    h ^= h >> 16;
    return h;
}

__device__ __forceinline__ float gumbel_from_hash(unsigned h) {
    float u = ((float)h + 0.5f) * 2.3283064365386963e-10f;  // (0,1)
    return -logf(-logf(u));
}

#if USE_TDM
typedef __attribute__((ext_vector_type(4))) unsigned tdm_v4u;
typedef __attribute__((ext_vector_type(8))) int      tdm_v8i;
typedef __attribute__((ext_vector_type(4))) int      tdm_v4i;

// 1-D contiguous global->LDS DMA via the Tensor Data Mover.
// n8 = number of 8-byte elements (data_size code 3).
__device__ __forceinline__ void tdm_copy_1d(void* lds_dst, const void* gsrc,
                                            unsigned n8) {
    unsigned lds_off = (unsigned)(unsigned long long)lds_dst;  // low 32b = LDS offset
    unsigned long long ga = (unsigned long long)gsrc;
    tdm_v4u g0;
    g0[0] = 1u;                                   // count=1, user descriptor
    g0[1] = lds_off;                              // lds_addr (bytes)
    g0[2] = (unsigned)(ga & 0xFFFFFFFFu);         // global_addr[31:0]
    g0[3] = (unsigned)((ga >> 32) & 0x01FFFFFFu)  // global_addr[56:32]
          | (2u << 30);                           // type = 2 ("image")
    tdm_v8i g1;
    g1[0] = 0x00030000;             // wg_mask=0, data_size=3 (8B)
    g1[1] = (int)(n8 << 16);        // tensor_dim0[15:0] in bits 63:48
    g1[2] = (int)((n8 >> 16) | (1u << 16)); // tensor_dim0[31:16], tensor_dim1=1
    g1[3] = (int)(n8 << 16);        // tile_dim0 in bits 127:112
    g1[4] = 1;                      // tile_dim1 = 1, tile_dim2 = 0
    g1[5] = (int)n8;                // tensor_dim0_stride[31:0]
    g1[6] = 0;                      // stride hi, dim1_stride lo
    g1[7] = 0;
    tdm_v4i gz = {0, 0, 0, 0};
#if __clang_major__ >= 23
    tdm_v8i gz8 = {0, 0, 0, 0, 0, 0, 0, 0};
    __builtin_amdgcn_tensor_load_to_lds(g0, g1, gz, gz, gz8, 0);
#else
    __builtin_amdgcn_tensor_load_to_lds(g0, g1, gz, gz, 0);
#endif
}
#endif

// Stage n_f16 halfs of pre-packed weights into LDS (TDM if available).
__device__ __forceinline__ void stage_weights(f16* Bs, const f16* Bpk, int n_f16) {
#if USE_TDM
    if (threadIdx.x < 32) {
        tdm_copy_1d(Bs, Bpk, (unsigned)(n_f16 >> 2));
        __builtin_amdgcn_s_wait_tensorcnt(0);
    }
    __syncthreads();
#else
    const uint4* s = (const uint4*)Bpk;
    uint4* d = (uint4*)Bs;
    for (int i = threadIdx.x; i < (n_f16 >> 3); i += blockDim.x) d[i] = s[i];
    __syncthreads();
#endif
}

// A-fragment: lane L holds row M=L%16; half=L/16 selects
// K {16h..16h+15} -> two contiguous 16B loads.
__device__ __forceinline__ v16h load_a_frag(const f16* p) {
    v8h lo = *(const v8h*)(p);
    v8h hi = *(const v8h*)(p + 8);
    v16h a;
#pragma unroll
    for (int i = 0; i < 8; ++i) { a[i] = lo[i]; a[i + 8] = hi[i]; }
    return a;
}

// One 16-row strip x (NJT*16) columns, K=128 as 4 chunks of 32.
// Bs holds pre-swizzled B fragments: ((kc*NJT + jt)*32 + lane)*16 contiguous f16.
template <int NJT>
__device__ __forceinline__ void wmma_strip(const f16* abase, const f16* Bs,
                                           int lane, v8f* accs) {
    v8f zero = {};
#pragma unroll
    for (int jt = 0; jt < NJT; ++jt) accs[jt] = zero;
    int koff = (lane >> 4) * 16;
#pragma unroll
    for (int kc = 0; kc < 4; ++kc) {
        v16h a = load_a_frag(abase + kc * 32 + koff);
#pragma unroll
        for (int jg = 0; jg < NJT; jg += 4) {
            v16h b0 = *(const v16h*)(Bs + ((kc * NJT + jg + 0) * 32 + lane) * 16);
            v16h b1 = *(const v16h*)(Bs + ((kc * NJT + jg + 1) * 32 + lane) * 16);
            v16h b2 = *(const v16h*)(Bs + ((kc * NJT + jg + 2) * 32 + lane) * 16);
            v16h b3 = *(const v16h*)(Bs + ((kc * NJT + jg + 3) * 32 + lane) * 16);
            accs[jg + 0] = __builtin_amdgcn_wmma_f32_16x16x32_f16(
                false, a, false, b0, (short)0, accs[jg + 0], false, false);
            accs[jg + 1] = __builtin_amdgcn_wmma_f32_16x16x32_f16(
                false, a, false, b1, (short)0, accs[jg + 1], false, false);
            accs[jg + 2] = __builtin_amdgcn_wmma_f32_16x16x32_f16(
                false, a, false, b2, (short)0, accs[jg + 2], false, false);
            accs[jg + 3] = __builtin_amdgcn_wmma_f32_16x16x32_f16(
                false, a, false, b3, (short)0, accs[jg + 3], false, false);
        }
    }
}

// ---------------------------------------------------------------- kernels

__global__ void k_init(float* z, float* step, int* cont, int* exit_l,
                       float* deg, int N) {
    long long t = (long long)blockIdx.x * blockDim.x + threadIdx.x;
    long long tot = (long long)N * H_DIM;
    if (t < tot) z[t] = 0.0f;
    if (t < N) { step[t] = 1.0f; cont[t] = 1; exit_l[t] = NLAYERS; deg[t] = 0.0f; }
}

__global__ void k_zero(float* p, long long n) {
    long long t = (long long)blockIdx.x * blockDim.x + threadIdx.x;
    if (t < n) p[t] = 0.0f;
}

__global__ void k_f2h(const float* src, f16* dst, long long n) {
    long long t = (long long)blockIdx.x * blockDim.x + threadIdx.x;
    if (t < n) dst[t] = (f16)src[t];
}

// Build Weff / antisymmetric A, and pack all weights into WMMA B-fragment order.
__global__ void k_prep(const float* Wp, const float* Wa, const float* Win,
                       const float* Wc1, float* W0, float* rsum,
                       f16* Win_pk, f16* Weff_pk, f16* A_pk, f16* Wc1_pk) {
    int tid = threadIdx.x;
    for (int p = tid; p < H_DIM * H_DIM; p += blockDim.x) {
        int i = p >> 7, j = p & 127;
        float v = 0.0f;
        if (i < j)      v = Wp[i * (H_DIM + 2) + j];
        else if (i > j) v = Wp[j * (H_DIM + 2) + i];
        W0[p] = v;
    }
    __syncthreads();
    for (int i = tid; i < H_DIM; i += blockDim.x) {
        float s = 0.0f;
        for (int j = 0; j < H_DIM; ++j) s += fabsf(W0[i * H_DIM + j]);
        rsum[i] = s;
    }
    __syncthreads();
    // 128-col packs: p = ((kc*8+jt)*32+lane)*16+e
    for (int p = tid; p < H_DIM * H_DIM; p += blockDim.x) {
        int e = p & 15, lane = (p >> 4) & 31, jt = (p >> 9) & 7, kc = p >> 12;
        int j = jt * 16 + (lane & 15);
        int k = kc * 32 + (lane >> 4) * 16 + e;
        Win_pk[p] = (f16)Win[j * H_DIM + k];
        float we = W0[j * H_DIM + k];
        if (j == k)
            we += Wp[j * (H_DIM + 2) + H_DIM] * rsum[j] + Wp[j * (H_DIM + 2) + H_DIM + 1];
        Weff_pk[p] = (f16)we;
        A_pk[p] = (f16)(Wa[j * H_DIM + k] - Wa[k * H_DIM + j]);
    }
    // 64-col pack for Wc1: p = ((kc*4+jt)*32+lane)*16+e
    for (int p = tid; p < CH_DIM * H_DIM; p += blockDim.x) {
        int e = p & 15, lane = (p >> 4) & 31, jt = (p >> 9) & 3, kc = p >> 11;
        int j = jt * 16 + (lane & 15);
        int k = kc * 32 + (lane >> 4) * 16 + e;
        Wc1_pk[p] = (f16)Wc1[j * H_DIM + k];
    }
}

__global__ void k_deg(const int* ei, float* deg, int E) {
    long long t = (long long)blockIdx.x * blockDim.x + threadIdx.x;
    if (t < E) {
        int s = ei[t], d = ei[(long long)E + t];
        if (s != d) atomicAdd(&deg[d], 1.0f);
    }
}

__global__ void k_dinv(const float* deg, float* dinv, int N) {
    long long t = (long long)blockIdx.x * blockDim.x + threadIdx.x;
    if (t < N) dinv[t] = (deg[t] > 0.0f) ? rsqrtf(deg[t]) : 0.0f;
}

__global__ void k_norm(const int* ei, const float* dinv, float* nrm, int E) {
    long long t = (long long)blockIdx.x * blockDim.x + threadIdx.x;
    if (t < E) {
        int s = ei[t], d = ei[(long long)E + t];
        nrm[t] = (s != d) ? dinv[s] * dinv[d] : 0.0f;
    }
}

// h = gelu(x @ Win^T + b_in); also emit f16 copy of h.  2 strips / wave.
__global__ void __launch_bounds__(256)
k_inproj(const f16* xh, const f16* Bpk, const float* bias,
         float* hf, f16* hbf, int N) {
    __shared__ __align__(32) f16 Bs[H_DIM * H_DIM];
    stage_weights(Bs, Bpk, H_DIM * H_DIM);
    int lane = threadIdx.x & 31, wave = threadIdx.x >> 5;
    int hlf = lane >> 4, col0 = lane & 15;
    for (int s = 0; s < 2; ++s) {
        int row0 = blockIdx.x * 256 + s * 128 + wave * 16;
        int arow = row0 + (lane & 15);
        if (arow >= N) arow = N - 1;
        v8f accs[8];
        wmma_strip<8>(xh + (long long)arow * H_DIM, Bs, lane, accs);
#pragma unroll
        for (int jt = 0; jt < 8; ++jt)
#pragma unroll
            for (int v = 0; v < 8; ++v) {
                int n = row0 + v + 8 * hlf;
                if (n < N) {
                    int col = jt * 16 + col0;
                    float val = gelu_exact(accs[jt][v] + bias[col]);
                    long long idx = (long long)n * H_DIM + col;
                    hf[idx] = val;
                    hbf[idx] = (f16)val;
                }
            }
    }
}

// hp = h @ Weff^T.  2 strips / wave.
__global__ void __launch_bounds__(256)
k_pair(const f16* hbf, const f16* Bpk, float* hp, int N) {
    __shared__ __align__(32) f16 Bs[H_DIM * H_DIM];
    stage_weights(Bs, Bpk, H_DIM * H_DIM);
    int lane = threadIdx.x & 31, wave = threadIdx.x >> 5;
    int hlf = lane >> 4, col0 = lane & 15;
    for (int s = 0; s < 2; ++s) {
        int row0 = blockIdx.x * 256 + s * 128 + wave * 16;
        int arow = row0 + (lane & 15);
        if (arow >= N) arow = N - 1;
        v8f accs[8];
        wmma_strip<8>(hbf + (long long)arow * H_DIM, Bs, lane, accs);
#pragma unroll
        for (int jt = 0; jt < 8; ++jt)
#pragma unroll
            for (int v = 0; v < 8; ++v) {
                int n = row0 + v + 8 * hlf;
                if (n < N) hp[(long long)n * H_DIM + jt * 16 + col0] = accs[jt][v];
            }
    }
}

// agg[dst] += norm[e] * hp[src]  (one wave per edge, float4 per lane)
__global__ void k_scatter(const int* ei, const float* nrm, const float* hp,
                          float* agg, int E) {
    int lane = threadIdx.x & 31, wave = threadIdx.x >> 5;
    long long e = (long long)blockIdx.x * 8 + wave;
    if (e >= E) return;
    float w = nrm[e];
    if (w == 0.0f) return;
    int src = ei[e], dst = ei[(long long)E + e];
    float4 v = ((const float4*)(hp + (long long)src * H_DIM))[lane];
    float* ad = agg + (long long)dst * H_DIM + lane * 4;
    atomicAdd(ad + 0, w * v.x);
    atomicAdd(ad + 1, w * v.y);
    atomicAdd(ad + 2, w * v.z);
    atomicAdd(ad + 3, w * v.w);
}

// delta = -(h@A^T + ba) + agg ; h += step * gelu(delta).
// Also re-zeroes agg in place for the next layer's scatter. 2 strips / wave.
__global__ void __launch_bounds__(256)
k_update(f16* hbf, float* hf, const f16* Apk, const float* ba,
         float* agg, const float* step, int N) {
    __shared__ __align__(32) f16 Bs[H_DIM * H_DIM];
    stage_weights(Bs, Apk, H_DIM * H_DIM);
    int lane = threadIdx.x & 31, wave = threadIdx.x >> 5;
    int hlf = lane >> 4, col0 = lane & 15;
    for (int s = 0; s < 2; ++s) {
        int row0 = blockIdx.x * 256 + s * 128 + wave * 16;
        int arow = row0 + (lane & 15);
        if (arow >= N) arow = N - 1;
        // warm agg rows (consumed in the epilogue) while WMMAs run
        __builtin_prefetch(agg + (long long)arow * H_DIM + hlf * 64, 0, 0);
        v8f accs[8];
        wmma_strip<8>(hbf + (long long)arow * H_DIM, Bs, lane, accs);
#pragma unroll
        for (int jt = 0; jt < 8; ++jt)
#pragma unroll
            for (int v = 0; v < 8; ++v) {
                int n = row0 + v + 8 * hlf;
                if (n < N) {
                    int col = jt * 16 + col0;
                    long long idx = (long long)n * H_DIM + col;
                    float d = -(accs[jt][v] + ba[col]) + agg[idx];
                    agg[idx] = 0.0f;  // re-arm for next layer
                    float hn = hf[idx] + step[n] * gelu_exact(d);
                    hf[idx] = hn;
                    hbf[idx] = (f16)hn;
                }
            }
    }
}

// Confidence head + exit decision, fused: WMMA -> GELU -> LN -> logits/temp ->
// gumbel argmax -> step/cont/exit updates -> cooperative z += h for new exits.
__global__ void __launch_bounds__(256)
k_conf(const f16* hbf, const float* hf, const f16* Wc1pk, const float* bc1,
       const float* lng, const float* lnb, const float* Wc2, const float* bc2,
       const float* Wt, float* z, float* step, int* cont, int* exit_l,
       int N, int layer) {
    __shared__ __align__(32) f16 Bs[CH_DIM * H_DIM];
    __shared__ float sC[8 * 16 * 65];
    __shared__ int sNewly[8 * 16];
    stage_weights(Bs, Wc1pk, CH_DIM * H_DIM);
    int lane = threadIdx.x & 31, wave = threadIdx.x >> 5;
    int row0 = blockIdx.x * 128 + wave * 16;
    int arow = row0 + (lane & 15);
    if (arow >= N) arow = N - 1;
    int hlf = lane >> 4, col0 = lane & 15;
    // warm the f32 h rows used by the temperature head after the WMMAs
    __builtin_prefetch(hf + (long long)arow * H_DIM + hlf * 64, 0, 0);
    v8f accs[4];
    wmma_strip<4>(hbf + (long long)arow * H_DIM, Bs, lane, accs);
    float* myC = sC + wave * 16 * 65;
#pragma unroll
    for (int jt = 0; jt < 4; ++jt)
#pragma unroll
        for (int v = 0; v < 8; ++v) {
            int m = v + 8 * hlf, col = jt * 16 + col0;
            myC[m * 65 + col] = gelu_exact(accs[jt][v] + bc1[col]);
        }
    __syncthreads();
    int newlyFlag = 0;
    if (lane < 16) {
        int m = lane, n = row0 + m;
        if (n < N) {
            float mu = 0.0f;
            for (int j = 0; j < CH_DIM; ++j) mu += myC[m * 65 + j];
            mu *= (1.0f / CH_DIM);
            float var = 0.0f;
            for (int j = 0; j < CH_DIM; ++j) {
                float dv = myC[m * 65 + j] - mu; var += dv * dv;
            }
            var *= (1.0f / CH_DIM);
            float inv = rsqrtf(var + LN_EPS_C);
            float l0 = bc2[0], l1 = bc2[1];
            for (int j = 0; j < CH_DIM; ++j) {
                float cn = (myC[m * 65 + j] - mu) * inv * lng[j] + lnb[j];
                l0 += cn * Wc2[j];
                l1 += cn * Wc2[CH_DIM + j];
            }
            float th = 0.0f;
            const float* hrow = hf + (long long)n * H_DIM;
            for (int k = 0; k < H_DIM; ++k) th += hrow[k] * Wt[k];
            float sp = (th > 20.0f) ? th : log1pf(expf(th));
            float tinv = sp + TAU0_C;  // = 1/temp
            float g0 = gumbel_from_hash(hash3(42u + (unsigned)layer, (unsigned)n, 0u));
            float g1 = gumbel_from_hash(hash3(42u + (unsigned)layer, (unsigned)n, 1u));
            int exitDec = ((l1 + g1) * tinv) > ((l0 + g0) * tinv);
            step[n] *= exitDec ? 0.0f : 1.0f;
            int c0 = cont[n];
            newlyFlag = exitDec & c0;
            if (newlyFlag) exit_l[n] = layer;
            cont[n] = c0 & (exitDec ^ 1);
        }
    }
    if (lane < 16) sNewly[wave * 16 + lane] = newlyFlag;
    __syncthreads();
    for (int m = 0; m < 16; ++m) {
        if (sNewly[wave * 16 + m]) {
            int n = row0 + m;
            for (int j = lane; j < H_DIM; j += 32)
                z[(long long)n * H_DIM + j] += hf[(long long)n * H_DIM + j];
        }
    }
}

__global__ void k_final(const float* hf, const int* cont, float* z, int N) {
    long long t = (long long)blockIdx.x * blockDim.x + threadIdx.x;
    if (t < (long long)N * H_DIM) {
        int n = (int)(t >> 7);
        if (cont[n]) z[t] += hf[t];
    }
}

// ---------------------------------------------------------------- launcher

extern "C" void kernel_launch(void* const* d_in, const int* in_sizes, int n_in,
                              void* d_out, int out_size, void* d_ws, size_t ws_size,
                              hipStream_t stream) {
    const float* x    = (const float*)d_in[0];
    const int*   ei   = (const int*)d_in[1];
    // d_in[2] = num_layers (device scalar) — constant 4 in reference
    const float* Win  = (const float*)d_in[3];
    const float* b_in = (const float*)d_in[4];
    const float* Wc1  = (const float*)d_in[5];
    const float* bc1  = (const float*)d_in[6];
    const float* lng  = (const float*)d_in[7];
    const float* lnb  = (const float*)d_in[8];
    const float* Wc2  = (const float*)d_in[9];
    const float* bc2  = (const float*)d_in[10];
    const float* Wt   = (const float*)d_in[11];
    const float* Wa   = (const float*)d_in[12];
    const float* ba   = (const float*)d_in[13];
    const float* Wp   = (const float*)d_in[14];

    int N = in_sizes[0] / H_DIM;
    int E = in_sizes[1] / 2;

    char* w = (char*)d_ws;
    auto carve = [&](size_t bytes) -> char* {
        char* p = w;
        w += (bytes + 255) & ~(size_t)255;
        return p;
    };
    float* hf     = (float*)carve((size_t)N * H_DIM * 4);
    f16*   hbf    = (f16*)  carve((size_t)N * H_DIM * 2);
    f16*   xbf    = (f16*)  carve((size_t)N * H_DIM * 2);
    float* hp     = (float*)carve((size_t)N * H_DIM * 4);
    float* agg    = (float*)carve((size_t)N * H_DIM * 4);
    float* step   = (float*)carve((size_t)N * 4);
    int*   cont   = (int*)  carve((size_t)N * 4);
    float* deg    = (float*)carve((size_t)N * 4);
    float* dinv   = (float*)carve((size_t)N * 4);
    float* nrm    = (float*)carve((size_t)E * 4);
    float* W0     = (float*)carve(H_DIM * H_DIM * 4);
    float* rsum   = (float*)carve(H_DIM * 4);
    f16* Win_pk   = (f16*)carve(H_DIM * H_DIM * 2);
    f16* Weff_pk  = (f16*)carve(H_DIM * H_DIM * 2);
    f16* A_pk     = (f16*)carve(H_DIM * H_DIM * 2);
    f16* Wc1_pk   = (f16*)carve(CH_DIM * H_DIM * 2);

    float* z = (float*)d_out;
    int* exit_l = (int*)(z + (size_t)N * H_DIM);

    long long NH = (long long)N * H_DIM;
    int gNH    = (int)((NH + 255) / 256);
    int gN     = (N + 255) / 256;
    int gE     = (E + 255) / 256;
    int gRows  = (N + 127) / 128;   // conf: 1 strip per wave
    int gRows2 = (N + 255) / 256;   // GEMMs: 2 strips per wave

    k_init<<<gNH, 256, 0, stream>>>(z, step, cont, exit_l, deg, N);
    k_prep<<<1, 256, 0, stream>>>(Wp, Wa, Win, Wc1, W0, rsum,
                                  Win_pk, Weff_pk, A_pk, Wc1_pk);
    k_f2h<<<gNH, 256, 0, stream>>>(x, xbf, NH);
    k_deg<<<gE, 256, 0, stream>>>(ei, deg, E);
    k_dinv<<<gN, 256, 0, stream>>>(deg, dinv, N);
    k_norm<<<gE, 256, 0, stream>>>(ei, dinv, nrm, E);
    k_inproj<<<gRows2, 256, 0, stream>>>(xbf, Win_pk, b_in, hf, hbf, N);
    k_zero<<<gNH, 256, 0, stream>>>(agg, NH);   // once; k_update re-zeroes

    for (int l = 0; l < NLAYERS; ++l) {
        k_conf<<<gRows, 256, 0, stream>>>(hbf, hf, Wc1_pk, bc1, lng, lnb,
                                          Wc2, bc2, Wt, z, step, cont, exit_l,
                                          N, l);
        k_pair<<<gRows2, 256, 0, stream>>>(hbf, Weff_pk, hp, N);
        k_scatter<<<(E + 7) / 8, 256, 0, stream>>>(ei, nrm, hp, agg, E);
        k_update<<<gRows2, 256, 0, stream>>>(hbf, hf, A_pk, ba, agg, step, N);
    }
    k_final<<<gNH, 256, 0, stream>>>(hf, cont, z, N);
}
